// MultiHeadAttention_83915071029298
// MI455X (gfx1250) — matrix operand order repaired and checked by reading, not compiled, not run on hardware
//
#include <hip/hip_runtime.h>

// ---------------- problem constants ----------------
#define BATCH   4
#define SEQ     2048
#define EMBED   512
#define NHEADS  16
#define DH      32          // head dim
#define KCH     64          // keys per flash chunk
#define QW      32          // queries per wave

typedef __attribute__((ext_vector_type(16))) _Float16 v16h;
typedef __attribute__((ext_vector_type(8)))  float    v8f;
typedef __attribute__((__vector_size__(4 * sizeof(int)))) int vsi4;
typedef __attribute__((ext_vector_type(4))) unsigned int u32x4;
typedef __attribute__((ext_vector_type(8))) int i32x8;
typedef __attribute__((ext_vector_type(4))) int i32x4;

// ---- gfx1250 feature probes (fallbacks keep compile green) ----
#if defined(__has_builtin)
#if __has_builtin(__builtin_amdgcn_global_load_async_to_lds_b128)
#define HAVE_ASYNC_B128 1
#endif
#if __has_builtin(__builtin_amdgcn_s_wait_asynccnt)
#define HAVE_WAIT_ASYNC 1
#endif
#if __has_builtin(__builtin_amdgcn_tensor_load_to_lds)
#define HAVE_TDM 1
#endif
#if __has_builtin(__builtin_amdgcn_s_wait_tensorcnt)
#define HAVE_WAIT_TENSOR 1
#endif
#endif

__device__ inline void stage16(const _Float16* __restrict__ g, _Float16* l) {
#ifdef HAVE_ASYNC_B128
    __builtin_amdgcn_global_load_async_to_lds_b128(
        (__attribute__((address_space(1))) vsi4*)g,
        (__attribute__((address_space(3))) vsi4*)l, 0, 0);
#else
    *(float4*)l = *(const float4*)g;
#endif
}

__device__ inline void wait_async_zero() {
#ifdef HAVE_ASYNC_B128
#ifdef HAVE_WAIT_ASYNC
    __builtin_amdgcn_s_wait_asynccnt(0);
#else
    asm volatile("s_wait_asynccnt 0x0" ::: "memory");
#endif
#endif
}

#ifdef HAVE_TDM
// TDM: DMA one 64x32 fp16 tile (row stride 32 elems) global -> LDS.
// D# per CDNA5 ISA 8.3/8.4: count=1, type=2(image), data_size=1(2B),
// tensor_dim0=32, tensor_dim1=SEQ, tile_dim0=32, tile_dim1=64, dim0_stride=32.
__device__ inline void tdm_load_tile(const _Float16* g, _Float16* ldsdst) {
    unsigned lds_off = (unsigned)(size_t)(__attribute__((address_space(3))) _Float16*)ldsdst;
    unsigned long long ga = (unsigned long long)(size_t)g;
    u32x4 g0;
    g0[0] = 1u;                                   // count=1 (valid user D#)
    g0[1] = lds_off;                              // lds_addr (bytes)
    g0[2] = (unsigned)ga;                         // global_addr[31:0]
    g0[3] = (unsigned)(ga >> 32) | (2u << 30);    // global_addr[56:32] | type=2
    i32x8 g1;
    g1[0] = (int)(1u << 16);                      // data_size = 1 -> 2 bytes
    g1[1] = (int)(32u << 16);                     // tensor_dim0 = 32 (bit 48)
    g1[2] = (int)((unsigned)SEQ << 16);           // tensor_dim1 (bit 80)
    g1[3] = (int)(32u << 16);                     // tile_dim0 = 32 (bit 112)
    g1[4] = KCH;                                  // tile_dim1 = 64 (bit 128)
    g1[5] = DH;                                   // tensor_dim0_stride = 32 (bit 160)
    g1[6] = 0;
    g1[7] = 0;
    i32x4 z4 = {0, 0, 0, 0};
    i32x8 z8 = {0, 0, 0, 0, 0, 0, 0, 0};
    __builtin_amdgcn_tensor_load_to_lds(g0, g1, z4, z4, z8, 0);
}
#endif

__device__ inline void wait_tensor_zero() {
#ifdef HAVE_TDM
#ifdef HAVE_WAIT_TENSOR
    __builtin_amdgcn_s_wait_tensorcnt(0);
#else
    asm volatile("s_wait_tensorcnt 0x0" ::: "memory");
#endif
#endif
}

// ---- fragment builders (contiguous k = 16*hi + j, same for A and B) ----
__device__ inline v16h ld16h(const _Float16* p) {     // B frag: 16 contiguous
    v16h r;
#pragma unroll
    for (int j = 0; j < 16; ++j) r[j] = p[j];
    return r;
}
__device__ inline v16h ldA(const _Float16* row, int hi) { // A frag: 16 contiguous K
    v16h r;
#pragma unroll
    for (int j = 0; j < 16; ++j) r[j] = row[hi * 16 + j];
    return r;
}
__device__ inline float redmax16(float x) {
#pragma unroll
    for (int off = 1; off < 16; off <<= 1) x = fmaxf(x, __shfl_xor(x, off, 32));
    return x;
}
__device__ inline float redsum16(float x) {
#pragma unroll
    for (int off = 1; off < 16; off <<= 1) x += __shfl_xor(x, off, 32);
    return x;
}

// =====================================================================
// Kernel 1: per-head projection  y = x @ W.T   (fp32 in, fp16 out)
// =====================================================================
__global__ __launch_bounds__(256) void proj_kernel(const float* __restrict__ x,
                                                   const float* __restrict__ W,
                                                   _Float16* __restrict__ out) {
    int gid = blockIdx.x * 256 + threadIdx.x;
    int e = gid & 31;
    int s = (gid >> 5) & (SEQ - 1);
    int h = (gid >> 16) & (NHEADS - 1);
    int b = gid >> 20;
    const float* xr = x + ((size_t)(b * SEQ + s)) * EMBED + h * DH;
    const float* wr = W + e * DH;
    float acc = 0.f;
#pragma unroll
    for (int d = 0; d < DH; ++d) acc += xr[d] * wr[d];
    out[gid] = (_Float16)acc;
}

// =====================================================================
// Kernel 1b: fp32 -> fp16 convert (for Wo)
// =====================================================================
__global__ __launch_bounds__(256) void cvt_kernel(const float* __restrict__ src,
                                                  _Float16* __restrict__ dst, int n) {
    int i = blockIdx.x * 256 + threadIdx.x;
    if (i < n) dst[i] = (_Float16)src[i];
}

// =====================================================================
// Kernel 2: fused flash attention, double-buffered staging.
//   K: async global->LDS (ASYNCcnt), V: TDM tensor_load_to_lds (TENSORcnt),
//   next chunk's DMA issued before current chunk's math (pipelined).
//   Per chunk per wave: 8 score WMMAs + 8 PV WMMAs.
// =====================================================================
__global__ __launch_bounds__(128) void attn_kernel(const _Float16* __restrict__ qh,
                                                   const _Float16* __restrict__ kh,
                                                   const _Float16* __restrict__ vh,
                                                   _Float16* __restrict__ ctx16) {
    __shared__ _Float16 Kbuf[2][KCH][DH];       // raw [key][d]       2x4 KB
    __shared__ _Float16 Vraw[2][KCH][DH];       // raw [key][d]       2x4 KB
    __shared__ _Float16 VbufT[2][DH][KCH];      // [d][key]           2x4 KB
    __shared__ _Float16 Pbuf[4][QW][KCH];       // per-wave P tiles    16 KB

    const int tid  = threadIdx.x;
    const int lane = tid & 31;
    const int w    = tid >> 5;
    const int lo   = lane & 15;
    const int hi   = lane >> 4;

    const int b = blockIdx.z;
    const int h = blockIdx.y;
    const int sq0 = blockIdx.x * 128 + w * QW;

    const size_t headbase = ((size_t)(b * NHEADS + h)) * SEQ * DH;
    const float scale = 0.17677669529663688f;   // 1/sqrt(32)

    // staging: K via per-lane async b128; V via one TDM tile (wave 0)
    auto stage_chunk = [&](int buf, int kc) {
#pragma unroll
        for (int i = 0; i < 2; ++i) {
            int seg = tid + i * 128;             // 0..255
            int row = seg >> 2, c16 = seg & 3;   // 64 rows x 4 segs
            stage16(kh + headbase + (size_t)(kc + row) * DH + c16 * 8,
                    &Kbuf[buf][row][c16 * 8]);
        }
#ifdef HAVE_TDM
        if (w == 0)
            tdm_load_tile(vh + headbase + (size_t)kc * DH, &Vraw[buf][0][0]);
#else
#pragma unroll
        for (int i = 0; i < 2; ++i) {
            int seg = tid + i * 128;
            int row = seg >> 2, c16 = seg & 3;
            stage16(vh + headbase + (size_t)(kc + row) * DH + c16 * 8,
                    &Vraw[buf][row][c16 * 8]);
        }
#endif
    };

    // ---- Q fragments (two 16x32 A tiles, built once) ----
    v16h aq[2];
#pragma unroll
    for (int qt = 0; qt < 2; ++qt)
        aq[qt] = ldA(qh + headbase + (size_t)(sq0 + qt * 16 + lo) * DH, hi);

    // ---- flash state ----
    float m[2][8], l[2][8];
    v8f c[2][2];
#pragma unroll
    for (int qt = 0; qt < 2; ++qt)
#pragma unroll
        for (int r = 0; r < 8; ++r) { m[qt][r] = -1e30f; l[qt][r] = 0.f; }
#pragma unroll
    for (int qt = 0; qt < 2; ++qt) { c[qt][0] = (v8f){}; c[qt][1] = (v8f){}; }

    stage_chunk(0, 0);   // prologue: prefetch first chunk

    for (int kc = 0; kc < SEQ; kc += KCH) {
        const int buf = (kc >> 6) & 1;

        // ---- current chunk's DMA complete; all prior reads retired ----
        wait_async_zero();
#ifdef HAVE_TDM
        if (w == 0) wait_tensor_zero();
#endif
        __syncthreads();

        // ---- kick off next chunk's DMA into the other buffer ----
        if (kc + KCH < SEQ) stage_chunk(buf ^ 1, kc + KCH);

        // ---- transpose V tile LDS->LDS: Vraw[buf] -> VbufT[buf] ----
#pragma unroll
        for (int i = 0; i < 2; ++i) {
            int seg = tid + i * 128;
            int row = seg >> 2, c16 = seg & 3;
            const _Float16* vr = &Vraw[buf][row][c16 * 8];
#pragma unroll
            for (int u = 0; u < 8; ++u) VbufT[buf][c16 * 8 + u][row] = vr[u];
        }

        // ---- K B-fragments (shared by both query tiles) ----
        v16h bk[4];
#pragma unroll
        for (int kt = 0; kt < 4; ++kt)
            bk[kt] = ld16h(&Kbuf[buf][kt * 16 + lo][hi * 16]);

        // ---- scores + online softmax, one pass per query tile ----
#pragma unroll
        for (int qt = 0; qt < 2; ++qt) {
            v8f zero = {};
            v8f s0 = __builtin_amdgcn_wmma_f32_16x16x32_f16(false, aq[qt], false, bk[0], (short)0, zero, false, false);
            v8f s1 = __builtin_amdgcn_wmma_f32_16x16x32_f16(false, aq[qt], false, bk[1], (short)0, zero, false, false);
            v8f s2 = __builtin_amdgcn_wmma_f32_16x16x32_f16(false, aq[qt], false, bk[2], (short)0, zero, false, false);
            v8f s3 = __builtin_amdgcn_wmma_f32_16x16x32_f16(false, aq[qt], false, bk[3], (short)0, zero, false, false);
#pragma unroll
            for (int r = 0; r < 8; ++r) {
                float e0 = s0[r] * scale, e1 = s1[r] * scale;
                float e2 = s2[r] * scale, e3 = s3[r] * scale;
                float rm = redmax16(fmaxf(fmaxf(e0, e1), fmaxf(e2, e3)));
                float mn = fmaxf(m[qt][r], rm);
                float al = __expf(m[qt][r] - mn);
                float p0 = __expf(e0 - mn), p1 = __expf(e1 - mn);
                float p2 = __expf(e2 - mn), p3 = __expf(e3 - mn);
                float rs = redsum16((p0 + p1) + (p2 + p3));
                l[qt][r] = l[qt][r] * al + rs;
                c[qt][0][r] *= al;
                c[qt][1][r] *= al;
                m[qt][r] = mn;
                int rowm = qt * 16 + r + 8 * hi;
                Pbuf[w][rowm][lo]      = (_Float16)p0;
                Pbuf[w][rowm][16 + lo] = (_Float16)p1;
                Pbuf[w][rowm][32 + lo] = (_Float16)p2;
                Pbuf[w][rowm][48 + lo] = (_Float16)p3;
            }
        }

        __syncthreads();   // VbufT[buf] transpose visible to all waves

        // ---- P @ V : V B-fragments shared across query tiles ----
        v16h bv[2][2];
#pragma unroll
        for (int g = 0; g < 2; ++g)
#pragma unroll
            for (int t = 0; t < 2; ++t)
                bv[g][t] = ld16h(&VbufT[buf][t * 16 + lo][g * 32 + hi * 16]);
#pragma unroll
        for (int qt = 0; qt < 2; ++qt) {
#pragma unroll
            for (int g = 0; g < 2; ++g) {
                v16h ap = ldA(&Pbuf[w][qt * 16 + lo][g * 32], hi);
                c[qt][0] = __builtin_amdgcn_wmma_f32_16x16x32_f16(false, ap, false, bv[g][0], (short)0, c[qt][0], false, false);
                c[qt][1] = __builtin_amdgcn_wmma_f32_16x16x32_f16(false, ap, false, bv[g][1], (short)0, c[qt][1], false, false);
            }
        }
        // next iteration's top barrier protects buffer reuse
    }

    // ---- normalize, write FP16 ctx in concat layout [b][s][EMBED] ----
#pragma unroll
    for (int qt = 0; qt < 2; ++qt) {
#pragma unroll
        for (int r = 0; r < 8; ++r) {
            float inv = 1.0f / l[qt][r];
            int row = sq0 + qt * 16 + r + 8 * hi;
            _Float16* orow = ctx16 + ((size_t)b * SEQ + row) * EMBED + h * DH;
            orow[lo]      = (_Float16)(c[qt][0][r] * inv);
            orow[16 + lo] = (_Float16)(c[qt][1][r] * inv);
        }
    }
}

// =====================================================================
// Kernel 3: output projection  out = concat @ Wo.T + bo  (f16 WMMA, K=512)
// =====================================================================
__global__ __launch_bounds__(128) void outproj_kernel(const _Float16* __restrict__ ctx16,
                                                      const _Float16* __restrict__ Wo16,
                                                      const float* __restrict__ bo,
                                                      float* __restrict__ out) {
    const int lane = threadIdx.x & 31;
    const int w    = threadIdx.x >> 5;
    const int lo   = lane & 15;
    const int hi   = lane >> 4;

    const int rt = blockIdx.x * 4 + w;      // row tile over B*SEQ/16
    const int ct = blockIdx.y;              // col tile over EMBED/16

    v8f acc = {};
#pragma unroll 4
    for (int kc = 0; kc < EMBED; kc += 32) {
        v16h a  = ldA(ctx16 + ((size_t)rt * 16 + lo) * EMBED + kc, hi);
        v16h bm = ld16h(Wo16 + (size_t)(ct * 16 + lo) * EMBED + kc + hi * 16);
        acc = __builtin_amdgcn_wmma_f32_16x16x32_f16(false, a, false, bm,
                                                     (short)0, acc, false, false);
    }

    float bias = bo[ct * 16 + lo];
#pragma unroll
    for (int r = 0; r < 8; ++r)
        out[((size_t)rt * 16 + r + 8 * hi) * EMBED + ct * 16 + lo] = acc[r] + bias;
}

// =====================================================================
// launch
// =====================================================================
extern "C" void kernel_launch(void* const* d_in, const int* in_sizes, int n_in,
                              void* d_out, int out_size, void* d_ws, size_t ws_size,
                              hipStream_t stream) {
    const float* key   = (const float*)d_in[0];
    const float* query = (const float*)d_in[1];
    const float* value = (const float*)d_in[2];
    const float* Wq    = (const float*)d_in[3];
    const float* Wk    = (const float*)d_in[4];
    const float* Wv    = (const float*)d_in[5];
    const float* Wo    = (const float*)d_in[6];
    const float* bo    = (const float*)d_in[7];
    float* out = (float*)d_out;

    const size_t NE = (size_t)BATCH * NHEADS * SEQ * DH;   // 4,194,304
    _Float16* qh    = (_Float16*)d_ws;
    _Float16* kh    = qh + NE;
    _Float16* vh    = kh + NE;
    _Float16* ctx16 = vh + NE;                             // B*SEQ*EMBED == NE
    _Float16* Wo16  = ctx16 + NE;
    // total ws use: (4*NE + 512*512) * 2 bytes  ≈ 32.5 MB

    dim3 pgrid((unsigned)(NE / 256));
    proj_kernel<<<pgrid, 256, 0, stream>>>(query, Wq, qh);
    proj_kernel<<<pgrid, 256, 0, stream>>>(key,   Wk, kh);
    proj_kernel<<<pgrid, 256, 0, stream>>>(value, Wv, vh);
    cvt_kernel<<<dim3(EMBED * EMBED / 256), 256, 0, stream>>>(Wo, Wo16, EMBED * EMBED);

    attn_kernel<<<dim3(SEQ / 128, NHEADS, BATCH), 128, 0, stream>>>(qh, kh, vh, ctx16);

    outproj_kernel<<<dim3((BATCH * SEQ / 16) / 4, EMBED / 16), 128, 0, stream>>>(
        ctx16, Wo16, bo, out);
}
